// SupConLoss_83124797047543
// MI455X (gfx1250) — matrix operand order repaired and checked by reading, not compiled, not run on hardware
//
#include <hip/hip_runtime.h>
#include <hip/hip_bf16.h>

#define TEMP       0.5f
#define BASE_TEMP  0.07f

typedef __attribute__((ext_vector_type(16))) __bf16 v16bf;
typedef __attribute__((ext_vector_type(8)))  __bf16 v8bf;
typedef __attribute__((ext_vector_type(8)))  float  v8f;
typedef __attribute__((ext_vector_type(4)))  unsigned int u32x4;
typedef __attribute__((ext_vector_type(8)))  unsigned int u32x8;

constexpr int N_ROWS      = 8192;   // 2*bsz
constexpr int D           = 512;
constexpr int BSZ         = 4096;
constexpr int NCLS        = 100;
constexpr int N_SEG       = 4;      // column segments (grid.y)
constexpr int ROWS_PER_WG = 64;     // 4 waves x 16 rows
constexpr int COLS_TILE   = 64;     // 4 x 16-col WMMA subtiles per wave
constexpr int K_CHUNK     = 64;     // staged K per TDM chunk
constexpr int N_KC        = D / K_CHUNK;                    // 8
constexpr int N_CT        = (N_ROWS / N_SEG) / COLS_TILE;   // 32
constexpr int TOTAL_IT    = N_CT * N_KC;                    // 256

// workspace layout (bytes)
constexpr size_t OFF_HI   = 0;
constexpr size_t OFF_LO   = OFF_HI + (size_t)N_ROWS * D * 2;        //  8 MB
constexpr size_t OFF_KEYS = OFF_LO + (size_t)N_ROWS * D * 2;        // 16 MB
constexpr size_t OFF_PM   = OFF_KEYS + (size_t)BSZ * 8;
constexpr size_t OFF_PS   = OFF_PM + (size_t)N_SEG * N_ROWS * 4;
constexpr size_t OFF_PA   = OFF_PS + (size_t)N_SEG * N_ROWS * 4;
constexpr size_t OFF_PC   = OFF_PA + (size_t)N_SEG * N_ROWS * 4;
constexpr size_t OFF_RL   = OFF_PC + (size_t)N_SEG * N_ROWS * 4;

union V16U { v16bf v; v8bf h[2]; };

// ---------------------------------------------------------------------------
// TDM: issue one 2-D tile load (64 rows x 64 bf16) via the Tensor Data Mover.
// D# group0 carries lds/global addresses (+type=2); group1 is a shared
// template (data_size=2B, tensor 512x8192 stride 512, tile 64x64).
// Inline asm form is portable across both toolchains (different builtin arity).
// ---------------------------------------------------------------------------
__device__ __forceinline__ void tdm_load_tile(unsigned lds_off,
                                              const __bf16* g,
                                              u32x8 g1) {
  unsigned long long ga = (unsigned long long)(size_t)g;
  u32x4 g0;
  g0.x = 1u;                                                // count=1 (user D#)
  g0.y = lds_off;                                           // lds_addr
  g0.z = (unsigned)ga;                                      // global_addr[31:0]
  g0.w = ((unsigned)(ga >> 32) & 0x01FFFFFFu) | 0x80000000u; // [56:32] + type=2
  asm volatile("tensor_load_to_lds %0, %1" :: "s"(g0), "s"(g1) : "memory");
}

__device__ __forceinline__ u32x8 tdm_make_g1() {
  u32x8 g1;
  g1[0] = (1u << 16);                                  // data_size=1 -> 2 bytes
  g1[1] = ((unsigned)D & 0xFFFFu) << 16;               // tensor_dim0[15:0]
  g1[2] = ((unsigned)D >> 16) | (((unsigned)N_ROWS & 0xFFFFu) << 16);
  g1[3] = ((unsigned)N_ROWS >> 16) | ((unsigned)K_CHUNK << 16); // tile_dim0
  g1[4] = (unsigned)COLS_TILE;                         // tile_dim1 (tile_dim2=0)
  g1[5] = (unsigned)D;                                 // tensor_dim0_stride
  g1[6] = 0u;
  g1[7] = 0u;
  return g1;
}

__device__ __forceinline__ unsigned lds_off_of(const void* p) {
  // flat LDS address = {shared_aperture, offset} ; low 32 bits are LDS offset
  return (unsigned)(unsigned long long)(size_t)p;
}

// ---------------------------------------------------------------------------
// Kernel 1: top-5 peel-argmax keys per label row. mask[i,j] == (key_i==key_j).
// ---------------------------------------------------------------------------
__global__ void topk_keys_kernel(const float* __restrict__ labels,
                                 unsigned long long* __restrict__ keys) {
  int row = blockIdx.x * blockDim.x + threadIdx.x;
  if (row >= BSZ) return;
  const float* L = labels + (size_t)row * NCLS;
  float peeled[5] = {-1e30f, -1e30f, -1e30f, -1e30f, -1e30f};
  unsigned long long key = 0;
  for (int pass = 0; pass < 5; ++pass) {
    float maxv = -1e30f; int maxi = 0;
    for (int e = 0; e < NCLS; ++e) {
      float v = L[e];
      #pragma unroll
      for (int q = 0; q < 5; ++q)
        if (q < pass && v == peeled[q]) v = 0.0f;   // emulate lab*(lab!=rowmax)
      if (v > maxv) { maxv = v; maxi = e; }          // first-index argmax
    }
    peeled[pass] = maxv;
    key = key * 100ull + (unsigned long long)maxi;
  }
  keys[row] = key;
}

// ---------------------------------------------------------------------------
// Kernel 2: split fp32 features into bf16 hi + bf16 lo (bf16x3 decomposition)
// ---------------------------------------------------------------------------
__global__ void cvt_kernel(const float* __restrict__ f,
                           __bf16* __restrict__ hi, __bf16* __restrict__ lo) {
  int i = (blockIdx.x * blockDim.x + threadIdx.x) * 4;
  if (i >= N_ROWS * D) return;
  float4 v = *(const float4*)(f + i);
  float a[4] = {v.x, v.y, v.z, v.w};
  #pragma unroll
  for (int j = 0; j < 4; ++j) {
    __bf16 h = (__bf16)a[j];
    hi[i + j] = h;
    lo[i + j] = (__bf16)(a[j] - (float)h);
  }
}

// ---------------------------------------------------------------------------
// Kernel 3: fused Gram-matrix + online softmax + masked sums.
// 64 rows per WG (4 waves x 16 rows), wave tile 16x64 (4 accumulators).
// TDM double-buffered staging: chunk k+1 DMAs into LDS while WMMAs chew k.
// ---------------------------------------------------------------------------
__global__ __launch_bounds__(128) void supcon_gemm_kernel(
    const __bf16* __restrict__ hiF, const __bf16* __restrict__ loF,
    const unsigned long long* __restrict__ keys,
    float* __restrict__ pm, float* __restrict__ ps,
    float* __restrict__ pa, float* __restrict__ pc) {
  __shared__ __bf16 sAhi[2][ROWS_PER_WG * K_CHUNK];   // 2 x 8 KB
  __shared__ __bf16 sAlo[2][ROWS_PER_WG * K_CHUNK];   // 2 x 8 KB
  __shared__ __bf16 sBhi[2][COLS_TILE   * K_CHUNK];   // 2 x 8 KB
  __shared__ __bf16 sBlo[2][COLS_TILE   * K_CHUNK];   // 2 x 8 KB  => 64 KB

  const int tid = threadIdx.x;
  const int w   = tid >> 5;        // wave id 0..3
  const int l   = tid & 31;        // lane
  const int lh  = l >> 4;          // lane half
  const int ln  = l & 15;
  const int rowBase = blockIdx.x * ROWS_PER_WG;
  const int seg = blockIdx.y;
  const float invT = 1.0f / TEMP;

  const u32x8 g1 = tdm_make_g1();
  unsigned ldsA_hi[2] = { lds_off_of(&sAhi[0][0]), lds_off_of(&sAhi[1][0]) };
  unsigned ldsA_lo[2] = { lds_off_of(&sAlo[0][0]), lds_off_of(&sAlo[1][0]) };
  unsigned ldsB_hi[2] = { lds_off_of(&sBhi[0][0]), lds_off_of(&sBhi[1][0]) };
  unsigned ldsB_lo[2] = { lds_off_of(&sBlo[0][0]), lds_off_of(&sBlo[1][0]) };

  // keys for this lane's 8 accumulator rows (C/D layout: M = p + 8*lh)
  unsigned long long rkey[8];
  #pragma unroll
  for (int p = 0; p < 8; ++p)
    rkey[p] = keys[(rowBase + w * 16 + p + 8 * lh) & (BSZ - 1)];

  float m_[8], s_[8], A_[8], C_[8];
  #pragma unroll
  for (int p = 0; p < 8; ++p) {
    m_[p] = -__builtin_inff(); s_[p] = 0.f; A_[p] = 0.f; C_[p] = 0.f;
  }

  // prologue: DMA chunk 0 into buffer 0 (wave 0 drives the TDM)
  if (w == 0) {
    const size_t aoff = (size_t)rowBase * D;
    const size_t boff = (size_t)(seg * (N_ROWS / N_SEG)) * D;
    tdm_load_tile(ldsA_hi[0], hiF + aoff, g1);
    tdm_load_tile(ldsA_lo[0], loF + aoff, g1);
    tdm_load_tile(ldsB_hi[0], hiF + boff, g1);
    tdm_load_tile(ldsB_lo[0], loF + boff, g1);
  }

  for (int ct2 = 0; ct2 < N_CT; ++ct2) {
    const int colBase = seg * (N_ROWS / N_SEG) + ct2 * COLS_TILE;
    v8f acc[4] = {{}, {}, {}, {}};

    for (int kc = 0; kc < N_KC; ++kc) {
      const int it  = ct2 * N_KC + kc;
      const int buf = it & 1;
      if (w == 0) __builtin_amdgcn_s_wait_tensorcnt(0);  // chunk `it` landed
      __syncthreads();                                   // publish buffer `buf`

      // overlap: DMA chunk it+1 into the other buffer (its last readers
      // finished before the barrier above)
      if (w == 0 && it + 1 < TOTAL_IT) {
        const int nit  = it + 1;
        const int nbuf = nit & 1;
        const int nct  = nit / N_KC;
        const int nkc  = nit % N_KC;
        const size_t aoff = (size_t)rowBase * D + nkc * K_CHUNK;
        const size_t boff =
            (size_t)(seg * (N_ROWS / N_SEG) + nct * COLS_TILE) * D +
            nkc * K_CHUNK;
        tdm_load_tile(ldsA_hi[nbuf], hiF + aoff, g1);
        tdm_load_tile(ldsA_lo[nbuf], loF + aoff, g1);
        tdm_load_tile(ldsB_hi[nbuf], hiF + boff, g1);
        tdm_load_tile(ldsB_lo[nbuf], loF + boff, g1);
      }

      #pragma unroll
      for (int ks = 0; ks < K_CHUNK / 32; ++ks) {
        const int k0 = ks * 32;
        // A 16x32 bf16 fragment: lane row = ln, K = k0+16*lh..+15 contiguous
        const int arow = (w * 16 + ln) * K_CHUNK;
        V16U ah, al;
        ah.h[0] = *(const v8bf*)(&sAhi[buf][arow + k0 + lh * 16]);
        ah.h[1] = *(const v8bf*)(&sAhi[buf][arow + k0 + lh * 16 + 8]);
        al.h[0] = *(const v8bf*)(&sAlo[buf][arow + k0 + lh * 16]);
        al.h[1] = *(const v8bf*)(&sAlo[buf][arow + k0 + lh * 16 + 8]);
        // B 32x16 fragments: lane col = st*16+ln, K = k0+16*lh..+15 contiguous
        const int kb = k0 + lh * 16;
        v16bf bh[4], bl[4];
        #pragma unroll
        for (int st = 0; st < 4; ++st) {
          bh[st] = *(const v16bf*)(&sBhi[buf][(st * 16 + ln) * K_CHUNK + kb]);
          bl[st] = *(const v16bf*)(&sBlo[buf][(st * 16 + ln) * K_CHUNK + kb]);
        }
        // bf16x3: lo*hi + hi*lo + hi*hi ; interleave 4 accumulators for ILP
        #pragma unroll
        for (int st = 0; st < 4; ++st)
          acc[st] = __builtin_amdgcn_wmma_f32_16x16x32_bf16(
              false, al.v, false, bh[st], (short)0, acc[st], false, false);
        #pragma unroll
        for (int st = 0; st < 4; ++st)
          acc[st] = __builtin_amdgcn_wmma_f32_16x16x32_bf16(
              false, ah.v, false, bl[st], (short)0, acc[st], false, false);
        #pragma unroll
        for (int st = 0; st < 4; ++st)
          acc[st] = __builtin_amdgcn_wmma_f32_16x16x32_bf16(
              false, ah.v, false, bh[st], (short)0, acc[st], false, false);
      }
    }

    // online softmax + exact masked sums for this 16x64 tile
    #pragma unroll
    for (int st = 0; st < 4; ++st) {
      const int gcol = colBase + st * 16 + ln;
      const unsigned long long ckey = keys[gcol & (BSZ - 1)];
      #pragma unroll
      for (int p = 0; p < 8; ++p) {
        const int grow = rowBase + w * 16 + p + 8 * lh;
        const float d = acc[st][p] * invT;
        const float nm = fmaxf(m_[p], d);                 // rowmax incl diag
        const bool diag = (grow == gcol);
        s_[p] = s_[p] * __expf(m_[p] - nm) + (diag ? 0.0f : __expf(d - nm));
        m_[p] = nm;
        if (!diag && rkey[p] == ckey) { A_[p] += d; C_[p] += 1.0f; }
      }
    }
  }

  // reduce across the 16 lanes that share each row (xor 1,2,4,8 stays in half)
  #pragma unroll
  for (int p = 0; p < 8; ++p) {
    float m = m_[p], s = s_[p], A = A_[p], C = C_[p];
    #pragma unroll
    for (int off = 8; off >= 1; off >>= 1) {
      float om = __shfl_xor(m, off, 32);
      float os = __shfl_xor(s, off, 32);
      float nm = fmaxf(m, om);
      s = s * __expf(m - nm) + os * __expf(om - nm);
      m = nm;
      A += __shfl_xor(A, off, 32);
      C += __shfl_xor(C, off, 32);
    }
    if (ln == 0) {
      const int grow = rowBase + w * 16 + p + 8 * lh;
      const size_t idx = (size_t)seg * N_ROWS + grow;
      pm[idx] = m; ps[idx] = s; pa[idx] = A; pc[idx] = C;
    }
  }
}

// ---------------------------------------------------------------------------
// Kernel 4: merge segment partials -> per-row loss
// ---------------------------------------------------------------------------
__global__ void rowloss_kernel(const float* __restrict__ pm, const float* __restrict__ ps,
                               const float* __restrict__ pa, const float* __restrict__ pc,
                               float* __restrict__ rowloss) {
  int r = blockIdx.x * blockDim.x + threadIdx.x;
  if (r >= N_ROWS) return;
  float m = -__builtin_inff(), s = 0.f, A = 0.f, C = 0.f;
  for (int seg = 0; seg < N_SEG; ++seg) {
    size_t idx = (size_t)seg * N_ROWS + r;
    float om = pm[idx], os = ps[idx];
    float nm = fmaxf(m, om);
    s = s * __expf(m - nm) + os * __expf(om - nm);
    m = nm;
    A += pa[idx]; C += pc[idx];
  }
  // mean_log_prob_pos = A/C - m - log(sum exp);  loss = -(T/Tb)*that
  rowloss[r] = -(TEMP / BASE_TEMP) * (A / C - m - __logf(s));
}

// ---------------------------------------------------------------------------
// Kernel 5: deterministic mean over 8192 rows (single workgroup tree)
// ---------------------------------------------------------------------------
__global__ void mean_kernel(const float* __restrict__ rowloss, float* __restrict__ out) {
  __shared__ float red[256];
  float acc = 0.f;
  for (int r = threadIdx.x; r < N_ROWS; r += 256) acc += rowloss[r];
  red[threadIdx.x] = acc;
  __syncthreads();
  for (int off = 128; off > 0; off >>= 1) {
    if ((int)threadIdx.x < off) red[threadIdx.x] += red[threadIdx.x + off];
    __syncthreads();
  }
  if (threadIdx.x == 0) out[0] = red[0] / (float)N_ROWS;
}

// ---------------------------------------------------------------------------
extern "C" void kernel_launch(void* const* d_in, const int* in_sizes, int n_in,
                              void* d_out, int out_size, void* d_ws, size_t ws_size,
                              hipStream_t stream) {
  const float* features = (const float*)d_in[0];
  const float* labels   = (const float*)d_in[1];
  char* ws = (char*)d_ws;
  __bf16* hiF = (__bf16*)(ws + OFF_HI);
  __bf16* loF = (__bf16*)(ws + OFF_LO);
  unsigned long long* keys = (unsigned long long*)(ws + OFF_KEYS);
  float* pm = (float*)(ws + OFF_PM);
  float* ps = (float*)(ws + OFF_PS);
  float* pa = (float*)(ws + OFF_PA);
  float* pc = (float*)(ws + OFF_PC);
  float* rowloss = (float*)(ws + OFF_RL);

  topk_keys_kernel<<<BSZ / 128, 128, 0, stream>>>(labels, keys);
  cvt_kernel<<<(N_ROWS * D / 4) / 256, 256, 0, stream>>>(features, hiF, loF);
  dim3 grid(N_ROWS / ROWS_PER_WG, N_SEG);
  supcon_gemm_kernel<<<grid, 128, 0, stream>>>(hiF, loF, keys, pm, ps, pa, pc);
  rowloss_kernel<<<N_ROWS / 256, 256, 0, stream>>>(pm, ps, pa, pc, rowloss);
  mean_kernel<<<1, 256, 0, stream>>>(rowloss, d_out ? (float*)d_out : nullptr);
}
